// Hierarch_RNN_85048942395578
// MI455X (gfx1250) — matrix operand order repaired and checked by reading, not compiled, not run on hardware
//
#include <hip/hip_runtime.h>

// ---------------------------------------------------------------------------
// Hierarchical RNN (SegRNN-like) for MI455X / gfx1250.
// Core: fused GRU-cell GEMM using v_wmma_f32_16x16x32_bf16.
// Each wave owns a 32x16 output tile: 12 accumulators
// ({r,z,n} x {Wih@x, Whh@h} x 2 M-subtiles) -> 12 WMMAs per K=32 chunk
// against 6 B-fragment + 4 A-fragment loads (B reused across M-subtiles).
// Gate math fused in the epilogue. Templated on <MODE, D>, K-loop fully
// unrolled with immediate-offset b128 loads.
// ---------------------------------------------------------------------------

#define SEQL 720
#define PREDL 96
#define ENC 321
#define BATCH 32
#define NB (BATCH * ENC) // 10272; NB/32 = 321

typedef __attribute__((ext_vector_type(16))) __bf16 v16bf;
typedef __attribute__((ext_vector_type(8)))  __bf16 v8bf;
typedef __attribute__((ext_vector_type(8)))  float  v8f;

__device__ __forceinline__ float sigmoidf_(float x) {
  return 1.0f / (1.0f + expf(-x));
}

__device__ __forceinline__ v8f wmma_bf16(v16bf a, v16bf b, v8f c) {
  return __builtin_amdgcn_wmma_f32_16x16x32_bf16(false, a, false, b,
                                                 (short)0, c, false, false);
}

// A-fragment 16x32 bf16: per-lane two contiguous 16B chunks.
// lane<16: K = k0+0..7 / k0+16..23 ; lane>=16: K = k0+8..15 / k0+24..31
// (p is already offset by hh ? 8 : 0)
__device__ __forceinline__ v16bf load_a_frag(const __bf16* __restrict__ p,
                                             int k0) {
  v8bf lo = *reinterpret_cast<const v8bf*>(p + k0);
  v8bf hi = *reinterpret_cast<const v8bf*>(p + k0 + 16);
  v16bf a;
#pragma unroll
  for (int j = 0; j < 8; ++j) { a[j] = lo[j]; a[j + 8] = hi[j]; }
  return a;
}

// B-fragment 32x16 bf16: lane holds W row (= output column), same K order as A.
// (p is already offset by hh ? 8 : 0)
__device__ __forceinline__ v16bf load_b_frag(const __bf16* __restrict__ p,
                                             int k0) {
  v8bf lo = *reinterpret_cast<const v8bf*>(p + k0);
  v8bf hi = *reinterpret_cast<const v8bf*>(p + k0 + 16);
  v16bf b;
#pragma unroll
  for (int j = 0; j < 8; ++j) { b[j] = lo[j]; b[j + 8] = hi[j]; }
  return b;
}

// ---------------------------------------------------------------------------
// Fused GRU cell:  h_out = (1-z)*tanh(gi_n + r*gh_n) + z*h_old
//   gi = A_x @ Wih^T + bih,  gh = A_h @ Whh^T + bhh
// MODE 0 (scan step):  A_x row M -> xsrc[M, :],  h row = M
// MODE 1 (decoder):    A_x row M -> concat(pos[M & smask], chan[(M>>hshift)%ENC])
//                      h row = M >> hshift          (hshift = log2(S))
// grid: (D/128, Mrows/32);  block: 256 threads (8 waves, 32x16 tile / wave)
// ---------------------------------------------------------------------------
template <int MODE, int D>
__global__ __launch_bounds__(256) void gru_gemm_kernel(
    const __bf16* __restrict__ xsrc, const float* __restrict__ pos_tab,
    const float* __restrict__ chan_tab, const __bf16* __restrict__ h_in,
    const __bf16* __restrict__ Wih, const __bf16* __restrict__ Whh,
    const float* __restrict__ bih, const float* __restrict__ bhh,
    __bf16* __restrict__ h_out, int hshift, int smask) {
  const int wv = threadIdx.x >> 5;
  const int lane = threadIdx.x & 31;
  const int hh = lane >> 4;
  const int l16 = lane & 15;

  const int m0 = blockIdx.y * 32;
  const int c0 = (blockIdx.x * 8 + wv) * 16;
  const int col = c0 + l16;
  const int arow0 = m0 + l16;
  const int arow1 = m0 + 16 + l16;
  const int hrow0 = (MODE == 0) ? arow0 : (arow0 >> hshift);
  const int hrow1 = (MODE == 0) ? arow1 : (arow1 >> hshift);

  // per-lane base pointers (inner loop uses constant immediate offsets)
  const int ao = hh ? 8 : 0;
  const int kbo = hh ? 8 : 0;
  const __bf16* pAh0 = h_in + (size_t)hrow0 * D + ao;
  const __bf16* pAh1 = h_in + (size_t)hrow1 * D + ao;
  const __bf16* pAx0 = xsrc + (size_t)arow0 * D + ao; // MODE 0 only
  const __bf16* pAx1 = xsrc + (size_t)arow1 * D + ao; // MODE 0 only
  const __bf16* pBri = Wih + (size_t)col * D + kbo;
  const __bf16* pBzi = Wih + (size_t)(D + col) * D + kbo;
  const __bf16* pBni = Wih + (size_t)(2 * D + col) * D + kbo;
  const __bf16* pBrh = Whh + (size_t)col * D + kbo;
  const __bf16* pBzh = Whh + (size_t)(D + col) * D + kbo;
  const __bf16* pBnh = Whh + (size_t)(2 * D + col) * D + kbo;

  // decoder A_x gather parameters
  constexpr int half = D >> 1;
  const int s1a = arow0 & smask;
  const int s1b = arow1 & smask;
  const int ech0 = (MODE == 1) ? (hrow0 % ENC) : 0;
  const int ech1 = (MODE == 1) ? (hrow1 % ENC) : 0;

  auto load_ax_dec = [&](int k0, int s1, int ech) -> v16bf {
    const int base_k = k0 + ao;
    v16bf a;
#pragma unroll
    for (int j = 0; j < 16; ++j) {
      int k = base_k + j + ((j >= 8) ? 8 : 0);
      float v = (k < half) ? pos_tab[s1 * half + k]
                           : chan_tab[(size_t)ech * half + (k - half)];
      a[j] = (__bf16)v;
    }
    return a;
  };

  // bias init (column = l16-indexed, identical across the 8 acc VGPRs)
  const float bir = bih[col], biz = bih[D + col], bin = bih[2 * D + col];
  const float bhr = bhh[col], bhz = bhh[D + col], bhn = bhh[2 * D + col];
  v8f aRi0, aZi0, aNi0, aRh0, aZh0, aNh0;
  v8f aRi1, aZi1, aNi1, aRh1, aZh1, aNh1;
#pragma unroll
  for (int r = 0; r < 8; ++r) {
    aRi0[r] = bir; aZi0[r] = biz; aNi0[r] = bin;
    aRh0[r] = bhr; aZh0[r] = bhz; aNh0[r] = bhn;
    aRi1[r] = bir; aZi1[r] = biz; aNi1[r] = bin;
    aRh1[r] = bhr; aZh1[r] = bhz; aNh1[r] = bhn;
  }

  // --- fully unrolled K loop: 10 loads then 12 WMMAs per chunk ---
#pragma unroll
  for (int k0 = 0; k0 < D; k0 += 32) {
    v16bf ah0 = load_a_frag(pAh0, k0);
    v16bf ah1 = load_a_frag(pAh1, k0);
    v16bf ax0, ax1;
    if (MODE == 0) {
      ax0 = load_a_frag(pAx0, k0);
      ax1 = load_a_frag(pAx1, k0);
    } else {
      ax0 = load_ax_dec(k0, s1a, ech0);
      ax1 = load_ax_dec(k0, s1b, ech1);
    }
    v16bf bRi = load_b_frag(pBri, k0);
    v16bf bZi = load_b_frag(pBzi, k0);
    v16bf bNi = load_b_frag(pBni, k0);
    v16bf bRh = load_b_frag(pBrh, k0);
    v16bf bZh = load_b_frag(pBzh, k0);
    v16bf bNh = load_b_frag(pBnh, k0);

    aRi0 = wmma_bf16(ax0, bRi, aRi0);
    aRi1 = wmma_bf16(ax1, bRi, aRi1);
    aZi0 = wmma_bf16(ax0, bZi, aZi0);
    aZi1 = wmma_bf16(ax1, bZi, aZi1);
    aNi0 = wmma_bf16(ax0, bNi, aNi0);
    aNi1 = wmma_bf16(ax1, bNi, aNi1);
    aRh0 = wmma_bf16(ah0, bRh, aRh0);
    aRh1 = wmma_bf16(ah1, bRh, aRh1);
    aZh0 = wmma_bf16(ah0, bZh, aZh0);
    aZh1 = wmma_bf16(ah1, bZh, aZh1);
    aNh0 = wmma_bf16(ah0, bNh, aNh0);
    aNh1 = wmma_bf16(ah1, bNh, aNh1);
  }

  // epilogue: full GRU gate math; C/D layout: VGPR r -> M = m_base + r + 8*hh
#pragma unroll
  for (int sub = 0; sub < 2; ++sub) {
    const v8f& Ri = sub ? aRi1 : aRi0;
    const v8f& Zi = sub ? aZi1 : aZi0;
    const v8f& Ni = sub ? aNi1 : aNi0;
    const v8f& Rh = sub ? aRh1 : aRh0;
    const v8f& Zh = sub ? aZh1 : aZh0;
    const v8f& Nh = sub ? aNh1 : aNh0;
#pragma unroll
    for (int r = 0; r < 8; ++r) {
      const int row = m0 + sub * 16 + r + (hh ? 8 : 0);
      const int hr = (MODE == 0) ? row : (row >> hshift);
      float rg = sigmoidf_(Ri[r] + Rh[r]);
      float zg = sigmoidf_(Zi[r] + Zh[r]);
      float ng = tanhf(Ni[r] + rg * Nh[r]);
      float hold = (float)h_in[(size_t)hr * D + col];
      float hnew = (1.0f - zg) * ng + zg * hold;
      h_out[(size_t)row * D + col] = (__bf16)hnew;
    }
  }
}

// ---------------------------------------------------------------------------
// Value embedding: xe[(seg - seg0)*NB + n, c] = relu(b[c] + sum_j xseg[n,j]*W[c,j])
// xseg[n,j] = x[b, seg*s + j, e] - x[b, SEQ-1, e],  n = b*ENC + e
// grid: (NB, nseg), block 128
// ---------------------------------------------------------------------------
__global__ __launch_bounds__(128) void embed_kernel(
    const float* __restrict__ x, const float* __restrict__ Wemb,
    const float* __restrict__ bemb, __bf16* __restrict__ xe, int d, int s,
    int seg0) {
  __shared__ float xs[64];
  const int n = blockIdx.x;
  const int seg = seg0 + blockIdx.y;
  const int b = n / ENC, e = n % ENC;
  const float* xb = x + ((size_t)b * SEQL) * ENC + e;
  if (threadIdx.x < s) {
    float last = xb[(size_t)(SEQL - 1) * ENC];
    xs[threadIdx.x] = xb[(size_t)(seg * s + threadIdx.x) * ENC] - last;
  }
  __syncthreads();
  for (int c = threadIdx.x; c < d; c += 128) {
    float acc = bemb[c];
#pragma unroll 4
    for (int j = 0; j < s; ++j) acc += xs[j] * Wemb[c * s + j];
    acc = fmaxf(acc, 0.0f);
    xe[((size_t)blockIdx.y * NB + n) * d + c] = (__bf16)acc;
  }
}

// ---------------------------------------------------------------------------
// Prediction head + scatter into the reference's permuted output layout:
//   val = hy[R] @ Wpred[j] + bpred[j];  L = R*s + j
//   L -> (b', e', p') over [B, ENC, PRED]; out[b', p', e'] += 0.5*val
// grid: S*NB blocks, block 64
// ---------------------------------------------------------------------------
__global__ __launch_bounds__(64) void pred_kernel(
    const __bf16* __restrict__ hy, const float* __restrict__ Wpred,
    const float* __restrict__ bpred, float* __restrict__ out, int d, int s) {
  __shared__ float hrow[512];
  const int R = blockIdx.x;
  for (int k = threadIdx.x; k < d; k += 64) hrow[k] = (float)hy[(size_t)R * d + k];
  __syncthreads();
  const int j = threadIdx.x;
  if (j < s) {
    float acc = bpred[j];
#pragma unroll 4
    for (int k = 0; k < d; ++k) acc += hrow[k] * Wpred[j * d + k];
    const int L = R * s + j;
    const int EP = ENC * PREDL;
    const int bp = L / EP, rem = L % EP;
    const int ep = rem / PREDL, pp = rem % PREDL;
    out[((size_t)bp * PREDL + pp) * ENC + ep] += 0.5f * acc;
  }
}

__global__ void init_out_kernel(const float* __restrict__ x,
                                float* __restrict__ out) {
  int i = blockIdx.x * 256 + threadIdx.x;
  if (i < BATCH * PREDL * ENC) {
    int e = i % ENC;
    int b = i / (PREDL * ENC);
    out[i] = x[((size_t)b * SEQL + (SEQL - 1)) * ENC + e]; // seq_last
  }
}

__global__ void f32_to_bf16_kernel(const float* __restrict__ src,
                                   __bf16* __restrict__ dst, int n) {
  int i = blockIdx.x * 256 + threadIdx.x;
  if (i < n) dst[i] = (__bf16)src[i];
}

__global__ void zero_bf16_kernel(__bf16* __restrict__ dst, int n) {
  int i = blockIdx.x * 256 + threadIdx.x;
  if (i < n) dst[i] = (__bf16)0.0f;
}

// ---------------------------------------------------------------------------
extern "C" void kernel_launch(void* const* d_in, const int* in_sizes, int n_in,
                              void* d_out, int out_size, void* d_ws,
                              size_t ws_size, hipStream_t stream) {
  const float* x      = (const float*)d_in[0];
  const float* Wemb0  = (const float*)d_in[1];
  const float* bemb0  = (const float*)d_in[2];
  const float* Wih0   = (const float*)d_in[3];
  const float* Whh0   = (const float*)d_in[4];
  const float* bih0   = (const float*)d_in[5];
  const float* bhh0   = (const float*)d_in[6];
  const float* Wpred0 = (const float*)d_in[7];
  const float* bpred0 = (const float*)d_in[8];
  const float* pos0   = (const float*)d_in[9];
  const float* chan0  = (const float*)d_in[10];
  const float* Wemb1  = (const float*)d_in[11];
  const float* bemb1  = (const float*)d_in[12];
  const float* Wih1   = (const float*)d_in[13];
  const float* Whh1   = (const float*)d_in[14];
  const float* bih1   = (const float*)d_in[15];
  const float* bhh1   = (const float*)d_in[16];
  const float* Wpred1 = (const float*)d_in[17];
  const float* bpred1 = (const float*)d_in[18];
  const float* pos1   = (const float*)d_in[19];
  const float* chan1  = (const float*)d_in[20];
  float* out = (float*)d_out;

  // workspace carve-up (bf16 elements), total ~67 MB
  __bf16* ws = (__bf16*)d_ws;
  size_t off = 0;
  __bf16* Wih0b = ws + off; off += (size_t)1536 * 512;
  __bf16* Whh0b = ws + off; off += (size_t)1536 * 512;
  __bf16* Wih1b = ws + off; off += (size_t)768 * 256;
  __bf16* Whh1b = ws + off; off += (size_t)768 * 256;
  __bf16* hA    = ws + off; off += (size_t)NB * 512;
  __bf16* hB    = ws + off; off += (size_t)NB * 512;
  __bf16* xe    = ws + off; off += (size_t)2 * NB * 512; // l0: [NB,512]; l1: [4,NB,256]
  __bf16* hy    = ws + off; off += (size_t)2 * NB * 512; // [S*NB, d] both layers

  auto cdiv = [](int a, int b) { return (a + b - 1) / b; };

  // one-time weight conversion to bf16
  f32_to_bf16_kernel<<<cdiv(1536 * 512, 256), 256, 0, stream>>>(Wih0, Wih0b, 1536 * 512);
  f32_to_bf16_kernel<<<cdiv(1536 * 512, 256), 256, 0, stream>>>(Whh0, Whh0b, 1536 * 512);
  f32_to_bf16_kernel<<<cdiv(768 * 256, 256), 256, 0, stream>>>(Wih1, Wih1b, 768 * 256);
  f32_to_bf16_kernel<<<cdiv(768 * 256, 256), 256, 0, stream>>>(Whh1, Whh1b, 768 * 256);

  // out = seq_last  (2*seq_last accumulated over 2 layers, /LAYERS)
  init_out_kernel<<<cdiv(BATCH * PREDL * ENC, 256), 256, 0, stream>>>(x, out);

  // ---------------- Layer 0: d=512, s=48, T=15 steps, S=2 ----------------
  zero_bf16_kernel<<<cdiv(NB * 512, 256), 256, 0, stream>>>(hA, NB * 512);
  for (int t = 0; t < 15; ++t) {
    embed_kernel<<<dim3(NB, 1), 128, 0, stream>>>(x, Wemb0, bemb0, xe, 512, 48, t);
    const __bf16* hin = (t & 1) ? hB : hA;
    __bf16* hout = (t & 1) ? hA : hB;
    gru_gemm_kernel<0, 512><<<dim3(512 / 128, NB / 32), 256, 0, stream>>>(
        xe, nullptr, nullptr, hin, Wih0b, Whh0b, bih0, bhh0, hout, 0, 0);
  }
  // h_fin0 = hB (last step t=14 writes hB). Decoder: S=2 -> hshift=1.
  gru_gemm_kernel<1, 512><<<dim3(512 / 128, 2 * NB / 32), 256, 0, stream>>>(
      nullptr, pos0, chan0, hB, Wih0b, Whh0b, bih0, bhh0, hy, 1, 1);
  pred_kernel<<<2 * NB, 64, 0, stream>>>(hy, Wpred0, bpred0, out, 512, 48);

  // ---------------- Layer 1: d=256, s=24, 60 steps over segs 0..3, S=4 ----
  zero_bf16_kernel<<<cdiv(NB * 256, 256), 256, 0, stream>>>(hA, NB * 256);
  embed_kernel<<<dim3(NB, 4), 128, 0, stream>>>(x, Wemb1, bemb1, xe, 256, 24, 0);
  for (int t = 0; t < 60; ++t) {
    const __bf16* xsrc = xe + (size_t)(t & 3) * NB * 256;
    const __bf16* hin = (t & 1) ? hB : hA;
    __bf16* hout = (t & 1) ? hA : hB;
    gru_gemm_kernel<0, 256><<<dim3(256 / 128, NB / 32), 256, 0, stream>>>(
        xsrc, nullptr, nullptr, hin, Wih1b, Whh1b, bih1, bhh1, hout, 0, 0);
  }
  // h_fin1 = hA (last step t=59 writes hA). Decoder: S=4 -> hshift=2.
  gru_gemm_kernel<1, 256><<<dim3(256 / 128, 4 * NB / 32), 256, 0, stream>>>(
      nullptr, pos1, chan1, hA, Wih1b, Whh1b, bih1, bhh1, hy, 2, 3);
  pred_kernel<<<4 * NB, 64, 0, stream>>>(hy, Wpred1, bpred1, out, 256, 24);
}
